// ComplexAttention_13494787244237
// MI455X (gfx1250) — compile-verified, matrix operand order checked by reading, not run on hardware
//
#include <hip/hip_runtime.h>
#include <hip/hip_bf16.h>
#include <math.h>

#define S_DIM 1024
#define B_DIM 4
#define E_DIM 1024
#define H_DIM 16
#define M_DIM 128
#define V_DIM 64
#define SB_DIM (S_DIM * B_DIM)

typedef __attribute__((ext_vector_type(16))) _Float16 v16h;
typedef __attribute__((ext_vector_type(8)))  _Float16 v8h;
typedef __attribute__((ext_vector_type(8)))  float    v8f;

#define WMMA_F16(A, B, C) \
  __builtin_amdgcn_wmma_f32_16x16x32_f16(false, (A), false, (B), (short)0, (C), false, false)

// 16x32 f16 A/B fragment from two contiguous 8-half segments
// (segment 0 at K-base, segment 1 at K-base+16; lane group picks +0 / +8).
static __device__ inline v16h frag_h(const _Float16* __restrict__ s0,
                                     const _Float16* __restrict__ s1) {
  v8h a = *(const v8h*)s0;
  v8h b = *(const v8h*)s1;
  v16h r;
#pragma unroll
  for (int j = 0; j < 8; ++j) {
    r[j]     = a[j];
    r[j + 8] = b[j];
  }
  return r;
}

static __device__ inline float rowmax16(float v) {
#pragma unroll
  for (int m = 1; m < 16; m <<= 1) v = fmaxf(v, __shfl_xor(v, m, 32));
  return v;
}
static __device__ inline float rowsum16(float v) {
#pragma unroll
  for (int m = 1; m < 16; m <<= 1) v += __shfl_xor(v, m, 32);
  return v;
}

// ---------------------------------------------------------------------------
// Bulk f32 -> f16 conversion (one pass; amortizes all GEMM operand reuse).
// ---------------------------------------------------------------------------
__global__ void cvt_f16_kernel(const float* __restrict__ src,
                               _Float16* __restrict__ dst, int n) {
  int i = blockIdx.x * blockDim.x + threadIdx.x;
  if (i < n) dst[i] = (_Float16)src[i];
}

// wo (E,E) f32 -> wo^T (F,E) f16, so out-proj B-fragments load contiguously.
__global__ void transpose_wo(const float* __restrict__ wre,
                             const float* __restrict__ wim,
                             _Float16* __restrict__ tre,
                             _Float16* __restrict__ tim) {
  int idx = blockIdx.x * blockDim.x + threadIdx.x;  // f*E + e
  if (idx >= E_DIM * E_DIM) return;
  int f = idx >> 10;
  int e = idx & (E_DIM - 1);
  tre[idx] = (_Float16)wre[(size_t)e * E_DIM + f];
  tim[idx] = (_Float16)wim[(size_t)e * E_DIM + f];
}

// ---------------------------------------------------------------------------
// Complex projection: out[h, m, sb] = sum_e w[h,m,e] * x[sb,e]   (complex)
// One wave per 16x16 output tile, all-f16 operands, f32 accumulation.
// TRANSPOSED=0 -> store [b][h][s][m] (q,k);  TRANSPOSED=1 -> [b][h][m][s] (v^T)
// ---------------------------------------------------------------------------
template <int TRANSPOSED>
__global__ __launch_bounds__(32)
void proj_kernel(const _Float16* __restrict__ wre, const _Float16* __restrict__ wim,
                 const _Float16* __restrict__ xre, const _Float16* __restrict__ xim,
                 _Float16* __restrict__ ore, _Float16* __restrict__ oim, int mdim) {
  const int lane = threadIdx.x;
  const int col  = lane & 15;
  const int hig  = lane >> 4;
  const int ct   = blockIdx.x;  // sb tile
  const int mt   = blockIdx.y;  // m tile
  const int h    = blockIdx.z;

  const int m_row = mt * 16 + col;
  const int sb    = ct * 16 + col;

  const _Float16* wr_row = wre + ((size_t)h * mdim + m_row) * E_DIM;
  const _Float16* wi_row = wim + ((size_t)h * mdim + m_row) * E_DIM;
  const _Float16* xr_col = xre + (size_t)sb * E_DIM;
  const _Float16* xi_col = xim + (size_t)sb * E_DIM;

  v8f rr = {}, ii = {}, ri = {}, ir = {};
  for (int e0 = 0; e0 < E_DIM; e0 += 32) {
    const int ka = e0 + hig * 8;
    v16h Ar = frag_h(wr_row + ka, wr_row + ka + 16);
    v16h Ai = frag_h(wi_row + ka, wi_row + ka + 16);
    v16h Br = frag_h(xr_col + ka, xr_col + ka + 16);
    v16h Bi = frag_h(xi_col + ka, xi_col + ka + 16);
    rr = WMMA_F16(Ar, Br, rr);
    ii = WMMA_F16(Ai, Bi, ii);
    ri = WMMA_F16(Ar, Bi, ri);
    ir = WMMA_F16(Ai, Br, ir);
  }

  const int s = sb >> 2;           // sb / B_DIM
  const int b = sb & (B_DIM - 1);  // sb % B_DIM
#pragma unroll
  for (int r = 0; r < 8; ++r) {
    const int mrow = mt * 16 + r + hig * 8;
    const float re = rr[r] - ii[r];
    const float im = ri[r] + ir[r];
    size_t idx;
    if (TRANSPOSED)
      idx = (((size_t)b * H_DIM + h) * mdim + mrow) * S_DIM + s;
    else
      idx = (((size_t)b * H_DIM + h) * S_DIM + s) * mdim + mrow;
    ore[idx] = (_Float16)re;
    oim[idx] = (_Float16)im;
  }
}

// ---------------------------------------------------------------------------
// Flash attention, complex scores: amp = |q k^T| / sqrt(M), causal mask,
// online softmax, upd = P V (complex). One wave per (b, h, 16-query tile).
// t processed in 32-column chunks so P@V WMMAs use the full K=32.
// ---------------------------------------------------------------------------
__global__ __launch_bounds__(32)
void attn_kernel(const _Float16* __restrict__ qre, const _Float16* __restrict__ qim,
                 const _Float16* __restrict__ kre, const _Float16* __restrict__ kim,
                 const _Float16* __restrict__ vre, const _Float16* __restrict__ vim,
                 _Float16* __restrict__ ure, _Float16* __restrict__ uim) {
  __shared__ __align__(16) _Float16 lds_p[16 * 32];  // 16 rows x 32 t-cols

  const int lane = threadIdx.x;
  const int col  = lane & 15;
  const int hig  = lane >> 4;
  const int st   = blockIdx.x;  // query tile
  const int h    = blockIdx.y;
  const int b    = blockIdx.z;

  const size_t bh   = (size_t)b * H_DIM + h;
  const int    srow = st * 16 + col;

  // Cache q fragments for this row tile (M=128 -> 4 K-steps, re+im).
  v16h qfr[4], qfi[4];
#pragma unroll
  for (int km = 0; km < 4; ++km) {
    const _Float16* pr = qre + (bh * S_DIM + srow) * M_DIM + km * 32 + hig * 8;
    const _Float16* pi = qim + (bh * S_DIM + srow) * M_DIM + km * 32 + hig * 8;
    qfr[km] = frag_h(pr, pr + 16);
    qfi[km] = frag_h(pi, pi + 16);
  }

  v8f accre[4], accim[4];
  {
    v8f z = {};
#pragma unroll
    for (int c = 0; c < 4; ++c) { accre[c] = z; accim[c] = z; }
  }
  float mrun[8], lrun[8];
#pragma unroll
  for (int r = 0; r < 8; ++r) { mrun[r] = -__builtin_inff(); lrun[r] = 0.f; }

  const float scl = 0.08838834764831845f;  // 1/sqrt(128)

  for (int tc = 0; tc <= st; tc += 2) {
    // ---- two score tiles (t columns tc*16 .. tc*16+31) ----
    float amp[2][8];
#pragma unroll
    for (int sub = 0; sub < 2; ++sub) {
      const int tt   = tc + sub;
      const int trow = tt * 16 + col;  // always < S_DIM
      v8f prr = {}, pii = {}, pri = {}, pir = {};
#pragma unroll
      for (int km = 0; km < 4; ++km) {
        const _Float16* kr = kre + (bh * S_DIM + trow) * M_DIM + km * 32 + hig * 8;
        const _Float16* ki = kim + (bh * S_DIM + trow) * M_DIM + km * 32 + hig * 8;
        if (sub == 0 && tc + 2 <= st) {  // prefetch next chunk's K rows
          __builtin_prefetch(kr + 32 * M_DIM, 0, 1);
          __builtin_prefetch(ki + 32 * M_DIM, 0, 1);
        }
        v16h Kr = frag_h(kr, kr + 16);
        v16h Ki = frag_h(ki, ki + 16);
        prr = WMMA_F16(qfr[km], Kr, prr);
        pii = WMMA_F16(qfi[km], Ki, pii);
        pri = WMMA_F16(qfr[km], Ki, pri);
        pir = WMMA_F16(qfi[km], Kr, pir);
      }
#pragma unroll
      for (int r = 0; r < 8; ++r) {
        const float re = prr[r] - pii[r];
        const float im = pri[r] + pir[r];
        const float a  = sqrtf(re * re + im * im) * scl;
        const int row  = st * 16 + r + hig * 8;
        amp[sub][r] = (trow <= row) ? a : -__builtin_inff();  // covers tt > st too
      }
    }

    // ---- online softmax over the 32 columns ----
    float p0[8], p1[8], scale[8];
#pragma unroll
    for (int r = 0; r < 8; ++r) {
      const float rm = fmaxf(rowmax16(amp[0][r]), rowmax16(amp[1][r]));
      const float mn = fmaxf(mrun[r], rm);
      const float sc = __expf(mrun[r] - mn);
      const float e0 = __expf(amp[0][r] - mn);
      const float e1 = __expf(amp[1][r] - mn);
      const float rs = rowsum16(e0) + rowsum16(e1);
      lrun[r]  = lrun[r] * sc + rs;
      mrun[r]  = mn;
      scale[r] = sc;
      p0[r] = e0;
      p1[r] = e1;
    }
#pragma unroll
    for (int c = 0; c < 4; ++c)
#pragma unroll
      for (int r = 0; r < 8; ++r) {
        accre[c][r] *= scale[r];
        accim[c][r] *= scale[r];
      }

    // ---- C-layout P (16x32) -> LDS -> full-K A fragment ----
    __syncthreads();
#pragma unroll
    for (int r = 0; r < 8; ++r) {
      lds_p[(r + hig * 8) * 32 + col]      = (_Float16)p0[r];
      lds_p[(r + hig * 8) * 32 + 16 + col] = (_Float16)p1[r];
    }
    __syncthreads();

    const _Float16* lp = &lds_p[col * 32 + hig * 8];
    v16h Pa = frag_h(lp, lp + 16);

    // ---- P @ V (re/im): 8 full-K WMMAs per 32-column chunk ----
#pragma unroll
    for (int c = 0; c < 4; ++c) {
      const int vc = c * 16 + col;
      const _Float16* vr = vre + (bh * V_DIM + vc) * S_DIM + tc * 16 + hig * 8;
      const _Float16* vi = vim + (bh * V_DIM + vc) * S_DIM + tc * 16 + hig * 8;
      v16h Vr = frag_h(vr, vr + 16);
      v16h Vi = frag_h(vi, vi + 16);
      accre[c] = WMMA_F16(Pa, Vr, accre[c]);
      accim[c] = WMMA_F16(Pa, Vi, accim[c]);
    }
  }

  // ---- normalize and store upd as f16 in (s, b, h*V + vc) layout ----
#pragma unroll
  for (int r = 0; r < 8; ++r) {
    const float inv = 1.0f / lrun[r];
    const int   s   = st * 16 + r + hig * 8;
#pragma unroll
    for (int c = 0; c < 4; ++c) {
      const int vc = c * 16 + col;
      const size_t idx = ((size_t)s * B_DIM + b) * E_DIM + h * V_DIM + vc;
      ure[idx] = (_Float16)(accre[c][r] * inv);
      uim[idx] = (_Float16)(accim[c][r] * inv);
    }
  }
}

// ---------------------------------------------------------------------------
// out[sb,f] = sum_e upd[sb,e] * wo[e,f] + x[sb,f]   (complex, f32 result)
// ---------------------------------------------------------------------------
__global__ __launch_bounds__(32)
void outproj_kernel(const _Float16* __restrict__ ure, const _Float16* __restrict__ uim,
                    const _Float16* __restrict__ wtre, const _Float16* __restrict__ wtim,
                    const float* __restrict__ xre, const float* __restrict__ xim,
                    float* __restrict__ out) {
  const int lane = threadIdx.x;
  const int col  = lane & 15;
  const int hig  = lane >> 4;
  const int it   = blockIdx.x;  // sb tile
  const int ft   = blockIdx.y;  // f tile

  const int sb = it * 16 + col;
  const int f  = ft * 16 + col;
  const _Float16* ar = ure + (size_t)sb * E_DIM;
  const _Float16* ai = uim + (size_t)sb * E_DIM;
  const _Float16* br = wtre + (size_t)f * E_DIM;
  const _Float16* bi = wtim + (size_t)f * E_DIM;

  v8f rr = {}, ii = {}, ri = {}, ir = {};
  for (int e0 = 0; e0 < E_DIM; e0 += 32) {
    const int ka = e0 + hig * 8;
    v16h Ar = frag_h(ar + ka, ar + ka + 16);
    v16h Ai = frag_h(ai + ka, ai + ka + 16);
    v16h Br = frag_h(br + ka, br + ka + 16);
    v16h Bi = frag_h(bi + ka, bi + ka + 16);
    rr = WMMA_F16(Ar, Br, rr);
    ii = WMMA_F16(Ai, Bi, ii);
    ri = WMMA_F16(Ar, Bi, ri);
    ir = WMMA_F16(Ai, Br, ir);
  }

  float* out_re = out;
  float* out_im = out + (size_t)SB_DIM * E_DIM;
#pragma unroll
  for (int r = 0; r < 8; ++r) {
    const int row_sb = it * 16 + r + hig * 8;
    const size_t idx = (size_t)row_sb * E_DIM + f;
    out_re[idx] = (rr[r] - ii[r]) + xre[idx];
    out_im[idx] = (ri[r] + ir[r]) + xim[idx];
  }
}

// ---------------------------------------------------------------------------
extern "C" void kernel_launch(void* const* d_in, const int* in_sizes, int n_in,
                              void* d_out, int out_size, void* d_ws, size_t ws_size,
                              hipStream_t stream) {
  (void)in_sizes; (void)n_in; (void)out_size; (void)ws_size;

  const float* xre  = (const float*)d_in[0];
  const float* xim  = (const float*)d_in[1];
  const float* wqre = (const float*)d_in[2];
  const float* wqim = (const float*)d_in[3];
  const float* wkre = (const float*)d_in[4];
  const float* wkim = (const float*)d_in[5];
  const float* wvre = (const float*)d_in[6];
  const float* wvim = (const float*)d_in[7];
  const float* wore = (const float*)d_in[8];
  const float* woim = (const float*)d_in[9];

  char*  ws = (char*)d_ws;
  size_t o  = 0;
  auto halloc = [&](size_t n) { _Float16* p = (_Float16*)(ws + o); o += n * sizeof(_Float16); return p; };

  const size_t QK = (size_t)B_DIM * H_DIM * S_DIM * M_DIM;  // q/k per component
  const size_t VT = (size_t)B_DIM * H_DIM * V_DIM * S_DIM;  // v^T per component
  const size_t UE = (size_t)SB_DIM * E_DIM;                 // upd / x per component
  const size_t WO = (size_t)E_DIM * E_DIM;                  // wo^T per component
  const size_t WQ = (size_t)H_DIM * M_DIM * E_DIM;          // wq/wk per component
  const size_t WV = (size_t)H_DIM * V_DIM * E_DIM;          // wv per component

  _Float16* qre  = halloc(QK);  _Float16* qim  = halloc(QK);
  _Float16* kre  = halloc(QK);  _Float16* kim  = halloc(QK);
  _Float16* vtre = halloc(VT);  _Float16* vtim = halloc(VT);
  _Float16* upre = halloc(UE);  _Float16* upim = halloc(UE);
  _Float16* wtre = halloc(WO);  _Float16* wtim = halloc(WO);
  _Float16* xhre = halloc(UE);  _Float16* xhim = halloc(UE);
  _Float16* qwre = halloc(WQ);  _Float16* qwim = halloc(WQ);
  _Float16* kwre = halloc(WQ);  _Float16* kwim = halloc(WQ);
  _Float16* vwre = halloc(WV);  _Float16* vwim = halloc(WV);

  // One-time f32 -> f16 operand conversions.
  auto cvt = [&](const float* s, _Float16* d, size_t n) {
    cvt_f16_kernel<<<(int)((n + 255) / 256), 256, 0, stream>>>(s, d, (int)n);
  };
  cvt(xre, xhre, UE);  cvt(xim, xhim, UE);
  cvt(wqre, qwre, WQ); cvt(wqim, qwim, WQ);
  cvt(wkre, kwre, WQ); cvt(wkim, kwim, WQ);
  cvt(wvre, vwre, WV); cvt(wvim, vwim, WV);
  transpose_wo<<<(E_DIM * E_DIM + 255) / 256, 256, 0, stream>>>(wore, woim, wtre, wtim);

  dim3 gq(SB_DIM / 16, M_DIM / 16, H_DIM);
  proj_kernel<0><<<gq, 32, 0, stream>>>(qwre, qwim, xhre, xhim, qre, qim, M_DIM);
  proj_kernel<0><<<gq, 32, 0, stream>>>(kwre, kwim, xhre, xhim, kre, kim, M_DIM);
  dim3 gv(SB_DIM / 16, V_DIM / 16, H_DIM);
  proj_kernel<1><<<gv, 32, 0, stream>>>(vwre, vwim, xhre, xhim, vtre, vtim, V_DIM);

  dim3 ga(S_DIM / 16, H_DIM, B_DIM);
  attn_kernel<<<ga, 32, 0, stream>>>(qre, qim, kre, kim, vtre, vtim, upre, upim);

  dim3 go(SB_DIM / 16, E_DIM / 16);
  outproj_kernel<<<go, 32, 0, stream>>>(upre, upim, wtre, wtim, xre, xim, (float*)d_out);
}
